// BGCN_20598663152187
// MI455X (gfx1250) — matrix operand order repaired and checked by reading, not compiled, not run on hardware
//
#include <hip/hip_runtime.h>

#define N_NODES 50000
#define N_EDGES 800000
#define HID 96
#define OUTD 64
#define NL 3
#define PRIOR_SIGMA 0.1f

typedef __attribute__((ext_vector_type(16))) _Float16 v16h;
typedef __attribute__((ext_vector_type(8)))  _Float16 v8h;
typedef __attribute__((ext_vector_type(8)))  float    v8f;

__device__ __forceinline__ float softplusf(float x) {
    return (x > 20.f) ? x : log1pf(expf(x));
}

__device__ __forceinline__ float kl_elem(float mu, float sig) {
    const float ps = PRIOR_SIGMA;
    return logf(ps) - logf(sig) + (sig * sig + mu * mu) / (2.f * ps * ps) - 0.5f;
}

// ---------------- utility ----------------
__global__ void k_zero(float* p, int n) {
    int i = blockIdx.x * blockDim.x + threadIdx.x;
    if (i < n) p[i] = 0.f;
}

// ---------------- degrees & norms ----------------
__global__ void k_degree(const int* __restrict__ src, const int* __restrict__ dst,
                         float* deg_out, float* deg_in) {
    int e = blockIdx.x * blockDim.x + threadIdx.x;
    if (e < N_EDGES) {
        atomicAdd(deg_out + src[e], 1.f);
        atomicAdd(deg_in + dst[e], 1.f);
    }
}

__global__ void k_norm(const float* __restrict__ deg_out, const float* __restrict__ deg_in,
                       float* norm_out, float* norm_in) {
    int i = blockIdx.x * blockDim.x + threadIdx.x;
    if (i < N_NODES) {
        float a = deg_out[i], b = deg_in[i];
        norm_out[i] = (a > 0.f) ? rsqrtf(a < 1.f ? 1.f : a) : 0.f;
        norm_in[i]  = (b > 0.f) ? rsqrtf(b < 1.f ? 1.f : b) : 0.f;
    }
}

// ---------------- sample conv weights (pack WMMA B layout) + KL ----------------
// WcB half index: ((((l*6+ct)*3+kt)*32+lane)*16 + e)
//   k = kt*32 + kk ; lane = (j%16) + 16*(kk>=16) ; e = kk%16 ; ct = j/16
__global__ void k_sample_conv(const float* __restrict__ wmu, const float* __restrict__ wrho,
                              const float* __restrict__ bmu, const float* __restrict__ brho,
                              const float* __restrict__ epsw, const float* __restrict__ epsb,
                              _Float16* __restrict__ WcB, float* __restrict__ bc_s,
                              float* __restrict__ kl) {
    const int NW = NL * HID * HID;   // 27648
    const int NB = NL * HID;         // 288
    int i = blockIdx.x * blockDim.x + threadIdx.x;
    float term = 0.f;
    if (i < NW) {
        float mu = wmu[i];
        float sig = softplusf(wrho[i]);
        float w = mu + sig * epsw[i];
        int l = i / (HID * HID);
        int r = i % (HID * HID);
        int k = r / HID, j = r % HID;
        int kt = k >> 5, kk = k & 31;
        int ct = j >> 4, col = j & 15;
        int lane = col + ((kk & 16) ? 16 : 0);
        int e = kk & 15;
        WcB[((((l * 6 + ct) * 3 + kt) * 32 + lane) << 4) + e] = (_Float16)w;
        term = kl_elem(mu, sig);
    } else if (i < NW + NB) {
        int bi = i - NW;
        float mu = bmu[bi];
        float sig = softplusf(brho[bi]);
        bc_s[bi] = mu + sig * epsb[bi];
        term = kl_elem(mu, sig);
    }
    __shared__ float red[256];
    red[threadIdx.x] = term;
    __syncthreads();
    for (int s = 128; s > 0; s >>= 1) {
        if ((int)threadIdx.x < s) red[threadIdx.x] += red[threadIdx.x + s];
        __syncthreads();
    }
    if (threadIdx.x == 0 && red[0] != 0.f) atomicAdd(kl, red[0]);
}

// ---------------- sample lin (JK head) weights, no KL ----------------
__global__ void k_sample_lin(const float* __restrict__ wmu, const float* __restrict__ wrho,
                             const float* __restrict__ bmu, const float* __restrict__ brho,
                             const float* __restrict__ epsw, const float* __restrict__ epsb,
                             float* __restrict__ Wl_s, float* __restrict__ bl_s) {
    const int NW = NL * HID * OUTD;  // 18432
    const int NB = NL * OUTD;        // 192
    int i = blockIdx.x * blockDim.x + threadIdx.x;
    if (i < NW) {
        Wl_s[i] = wmu[i] + softplusf(wrho[i]) * epsw[i];
    } else if (i < NW + NB) {
        int bi = i - NW;
        bl_s[bi] = bmu[bi] + softplusf(brho[bi]) * epsb[bi];
    }
}

// ---------------- column sums of h (for node-mean of JK head) ----------------
__global__ void k_colsum(const float* __restrict__ h, float* __restrict__ cs) {
    int j = threadIdx.x;             // 0..95, 3 waves
    int r0 = blockIdx.x * 128;
    int r1 = r0 + 128; if (r1 > N_NODES) r1 = N_NODES;
    float s = 0.f;
    for (int r = r0; r < r1; ++r) s += h[r * HID + j];
    atomicAdd(cs + j, s);
}

// ---------------- edge gather/scatter: m[dst] += h[src]*norm_out[src] ----------------
__global__ void k_scatter(const float* __restrict__ h, const float* __restrict__ norm_out,
                          const int* __restrict__ src, const int* __restrict__ dst,
                          float* __restrict__ m) {
    int t = blockIdx.x * blockDim.x + threadIdx.x;
    if (t >= N_EDGES * HID) return;
    int e = t / HID, j = t % HID;
    int s = src[e], d = dst[e];
    atomicAdd(m + d * HID + j, h[s * HID + j] * norm_out[s]);
}

// ---------------- scale by norm_in and convert to f16 ----------------
__global__ void k_scaleconv(const float* __restrict__ m, const float* __restrict__ norm_in,
                            _Float16* __restrict__ mh) {
    int t = blockIdx.x * blockDim.x + threadIdx.x;
    if (t < N_NODES * HID) mh[t] = (_Float16)(m[t] * norm_in[t / HID]);
}

// ---------------- WMMA GEMM: h = relu(m @ Wc + bc), [50000x96]x[96x96] ----------------
__global__ void k_gemm(const _Float16* __restrict__ mh, const _Float16* __restrict__ WcBl,
                       const float* __restrict__ bc, float* __restrict__ hout) {
    const int NT = (N_NODES / 16) * (HID / 16);  // 3125 * 6 = 18750 tiles, one wave each
    int wave = (blockIdx.x * blockDim.x + threadIdx.x) >> 5;
    int lane = threadIdx.x & 31;
    if (wave >= NT) return;
    int rt = wave / 6, ct = wave % 6;

    int arow = rt * 16 + (lane & 15);            // A: lanes 0-15 & 16-31 both hold M=0..15
    int koff = (lane & 16) ? 16 : 0;             // per-lane K sub-offset in each 32-K tile

    v8f acc = {};
    #pragma unroll
    for (int kt = 0; kt < 3; ++kt) {
        union { v16h v; v8h h2[2]; } A, B;
        const _Float16* ap = mh + arow * HID + kt * 32;
        A.h2[0] = *(const v8h*)(ap + koff);          // elements 0..7 : K = koff + 0..7
        A.h2[1] = *(const v8h*)(ap + 8 + koff);      // elements 8..15: K = koff + 8..15
        const _Float16* bp = WcBl + (((ct * 3 + kt) * 32 + lane) << 4);
        B.h2[0] = *(const v8h*)(bp);
        B.h2[1] = *(const v8h*)(bp + 8);
        acc = __builtin_amdgcn_wmma_f32_16x16x32_f16(
            /*neg_a=*/false, A.v, /*neg_b=*/false, B.v,
            /*c_mod=*/(short)0, acc, /*reuse_a=*/false, /*reuse_b=*/false);
    }

    int col = ct * 16 + (lane & 15);
    int rbase = rt * 16 + ((lane & 16) ? 8 : 0);
    float bias = bc[col];
    #pragma unroll
    for (int v = 0; v < 8; ++v) {
        float o = acc[v] + bias;
        hout[(rbase + v) * HID + col] = (o > 0.f) ? o : 0.f;
    }
}

// ---------------- head: out[j] = mean_l[(colsum/N) @ Wl + bl], out[64] = kl ----------------
__global__ void k_head(const float* __restrict__ cs, const float* __restrict__ Wl_s,
                       const float* __restrict__ bl_s, const float* __restrict__ kl,
                       float* __restrict__ out) {
    int j = threadIdx.x;
    if (j < OUTD) {
        float s = 0.f;
        const float invN = 1.f / (float)N_NODES;
        for (int l = 0; l < NL; ++l) {
            float t = bl_s[l * OUTD + j];
            const float* c = cs + l * HID;
            const float* W = Wl_s + l * HID * OUTD;
            for (int k = 0; k < HID; ++k)
                t += (c[k] * invN) * W[k * OUTD + j];
            s += t;
        }
        out[j] = s * (1.f / 3.f);
    } else if (j == OUTD) {
        out[OUTD] = kl[0];
    }
}

extern "C" void kernel_launch(void* const* d_in, const int* in_sizes, int n_in,
                              void* d_out, int out_size, void* d_ws, size_t ws_size,
                              hipStream_t stream) {
    (void)in_sizes; (void)n_in; (void)out_size; (void)ws_size;
    const float* feat  = (const float*)d_in[0];
    const float* cwmu  = (const float*)d_in[1];
    const float* cwrho = (const float*)d_in[2];
    const float* cbmu  = (const float*)d_in[3];
    const float* cbrho = (const float*)d_in[4];
    const float* cepsw = (const float*)d_in[5];
    const float* cepsb = (const float*)d_in[6];
    const float* lwmu  = (const float*)d_in[7];
    const float* lwrho = (const float*)d_in[8];
    const float* lbmu  = (const float*)d_in[9];
    const float* lbrho = (const float*)d_in[10];
    const float* lepsw = (const float*)d_in[11];
    const float* lepsb = (const float*)d_in[12];
    const int*   src   = (const int*)d_in[13];
    const int*   dst   = (const int*)d_in[14];
    float* out = (float*)d_out;

    char* ws = (char*)d_ws;
    size_t off = 0;
    auto take = [&](size_t bytes) -> char* {
        char* p = ws + off;
        off += (bytes + 255) & ~(size_t)255;
        return p;
    };
    float*    deg_out  = (float*)take((size_t)N_NODES * 4);
    float*    deg_in   = (float*)take((size_t)N_NODES * 4);
    float*    norm_out = (float*)take((size_t)N_NODES * 4);
    float*    norm_in  = (float*)take((size_t)N_NODES * 4);
    float*    h        = (float*)take((size_t)N_NODES * HID * 4);
    float*    m        = (float*)take((size_t)N_NODES * HID * 4);
    _Float16* mh       = (_Float16*)take((size_t)N_NODES * HID * 2);
    _Float16* WcB      = (_Float16*)take((size_t)NL * 6 * 3 * 32 * 16 * 2);
    float*    bc_s     = (float*)take((size_t)NL * HID * 4);
    float*    Wl_s     = (float*)take((size_t)NL * HID * OUTD * 4);
    float*    bl_s     = (float*)take((size_t)NL * OUTD * 4);
    float*    cs       = (float*)take((size_t)NL * HID * 4);
    float*    klbuf    = (float*)take(4);

    const int B = 256;
    // zero accumulators
    k_zero<<<(N_NODES + B - 1) / B, B, 0, stream>>>(deg_out, N_NODES);
    k_zero<<<(N_NODES + B - 1) / B, B, 0, stream>>>(deg_in, N_NODES);
    k_zero<<<1, B, 0, stream>>>(cs, NL * HID);
    k_zero<<<1, 32, 0, stream>>>(klbuf, 1);

    k_degree<<<(N_EDGES + B - 1) / B, B, 0, stream>>>(src, dst, deg_out, deg_in);
    k_norm<<<(N_NODES + B - 1) / B, B, 0, stream>>>(deg_out, deg_in, norm_out, norm_in);

    const int NSW = NL * HID * HID + NL * HID;
    k_sample_conv<<<(NSW + B - 1) / B, B, 0, stream>>>(cwmu, cwrho, cbmu, cbrho,
                                                       cepsw, cepsb, WcB, bc_s, klbuf);
    const int NSL = NL * HID * OUTD + NL * OUTD;
    k_sample_lin<<<(NSL + B - 1) / B, B, 0, stream>>>(lwmu, lwrho, lbmu, lbrho,
                                                      lepsw, lepsb, Wl_s, bl_s);

    const int NM = N_NODES * HID;
    const int NE = N_EDGES * HID;
    const int NTW = (N_NODES / 16) * (HID / 16);       // 18750 waves
    const int gemmBlocks = (NTW * 32 + B - 1) / B;
    for (int l = 0; l < NL; ++l) {
        const float* hl = (l == 0) ? feat : h;
        // colsum of this layer's input (JK head, node-mean commutes with linear map)
        k_colsum<<<(N_NODES + 127) / 128, HID, 0, stream>>>(hl, cs + l * HID);
        // aggregate messages
        k_zero<<<(NM + B - 1) / B, B, 0, stream>>>(m, NM);
        k_scatter<<<(NE + B - 1) / B, B, 0, stream>>>(hl, norm_out, src, dst, m);
        k_scaleconv<<<(NM + B - 1) / B, B, 0, stream>>>(m, norm_in, mh);
        // h = relu(m @ Wc + bc) via WMMA
        k_gemm<<<gemmBlocks, B, 0, stream>>>(mh, WcB + (size_t)l * 6 * 3 * 32 * 16,
                                             bc_s + l * HID, h);
    }

    k_head<<<1, 96, 0, stream>>>(cs, Wl_s, bl_s, klbuf, out);
}
